// CausalSelfAttention_60275571032748
// MI455X (gfx1250) — matrix operand order repaired and checked by reading, not compiled, not run on hardware
//
#include <hip/hip_runtime.h>

typedef __attribute__((ext_vector_type(16))) _Float16     v16h;
typedef __attribute__((ext_vector_type(8)))  float        v8f;
typedef __attribute__((ext_vector_type(8)))  unsigned int v8u;

#define TT 2048
#define CC 1024
#define NH 16
#define HD 64

__device__ __forceinline__ v8f wmma16(v16h a, v16h b, v8f c) {
  return __builtin_amdgcn_wmma_f32_16x16x32_f16(false, a, false, b, (short)0, c, false, false);
}

// A fragment (16x32 f16, M x K) from LDS row-major [row][k].
__device__ __forceinline__ v16h ldsA(const _Float16* s, int row0, int k0, int ld) {
  const int lane = threadIdx.x & 31;
  const int m = lane & 15, hf = lane >> 4;
  const _Float16* p = s + (row0 + m) * ld + k0 + hf * 16;
  v8u u;
#pragma unroll
  for (int i = 0; i < 4; ++i) u[i] = *(const unsigned int*)(p + 2 * i);
#pragma unroll
  for (int i = 0; i < 4; ++i) u[4 + i] = *(const unsigned int*)(p + 8 + 2 * i);
  return __builtin_bit_cast(v16h, u);
}

// B fragment (32x16 f16, K x N) from LDS stored as [n][k] (K-pairs contiguous).
__device__ __forceinline__ v16h ldsB(const _Float16* s, int n0, int k0, int ld) {
  const int lane = threadIdx.x & 31;
  const int n = lane & 15, hf = lane >> 4;
  const _Float16* p = s + (n0 + n) * ld + k0 + hf * 16;
  v8u u;
#pragma unroll
  for (int j = 0; j < 8; ++j) u[j] = *(const unsigned int*)(p + 2 * j);
  return __builtin_bit_cast(v16h, u);
}

// B fragment where LDS holds the matrix row-major as [k][n] (no transpose staged):
// gather two 16-bit elements per VGPR at k-stride = ld and pack.
__device__ __forceinline__ v16h ldsBt(const _Float16* s, int n0, int k0, int ld) {
  const int lane = threadIdx.x & 31;
  const int n = lane & 15, hf = lane >> 4;
  const unsigned short* p =
      (const unsigned short*)(s + (k0 + hf * 16) * ld + (n0 + n));
  v8u u;
#pragma unroll
  for (int j = 0; j < 8; ++j) {
    unsigned lo = p[(2 * j) * ld];
    unsigned hi = p[(2 * j + 1) * ld];
    u[j] = lo | (hi << 16);
  }
  return __builtin_bit_cast(v16h, u);
}

// CDNA5 async DMA: copy 4 bytes/lane global -> LDS, tracked by ASYNCcnt.
__device__ __forceinline__ void async_ld_b32(unsigned lds_off, const void* g) {
  asm volatile("global_load_async_to_lds_b32 %0, %1, off"
               :: "v"(lds_off), "v"((unsigned long long)(size_t)g)
               : "memory");
}
__device__ __forceinline__ void wait_async0() {
  asm volatile("s_wait_asynccnt 0x0" ::: "memory");
}

// ---------------- Kernel 1: qkv = x @ Wqkv + b, scatter to [B,H,T,D] f16 -----
__global__ __launch_bounds__(256) void csa_qkv_gemm(
    const float* __restrict__ x, const float* __restrict__ W,
    const float* __restrict__ bias,
    _Float16* __restrict__ qb, _Float16* __restrict__ kb, _Float16* __restrict__ vb)
{
  const int LDT = 34;
  __shared__ __align__(16) _Float16 As[128 * 34];
  __shared__ __align__(16) _Float16 Bs[128 * 34];
  const int gm0 = blockIdx.y * 128;
  const int gn0 = blockIdx.x * 128;
  const int tid = threadIdx.x;
  const int wave = tid >> 5, lane = tid & 31;
  const int wr = wave >> 1, wc = wave & 1;
  const int hf = lane >> 4, nn = lane & 15;

  v8f acc[2][4] = {};
  for (int k0 = 0; k0 < CC; k0 += 32) {
    if (k0 + 32 < CC) {  // cover next tile's latency (global_prefetch_b8)
      __builtin_prefetch(&x[(size_t)(gm0 + (tid >> 5)) * CC + k0 + 32 + (tid & 31)], 0, 0);
      __builtin_prefetch(&W[(size_t)(k0 + 32 + (tid >> 7)) * (3 * CC) + gn0 + (tid & 127)], 0, 0);
    }
#pragma unroll
    for (int i = 0; i < 16; ++i) {             // A tile 128x32, fp32 -> f16
      int idx = tid + i * 256;
      int r = idx >> 5, c = idx & 31;
      As[r * LDT + c] = (_Float16)x[(size_t)(gm0 + r) * CC + k0 + c];
    }
#pragma unroll
    for (int i = 0; i < 16; ++i) {             // B tile 32x128 stored [n][k]
      int idx = tid + i * 256;
      int rk = idx >> 7, cn = idx & 127;
      Bs[cn * LDT + rk] = (_Float16)W[(size_t)(k0 + rk) * (3 * CC) + gn0 + cn];
    }
    __syncthreads();
    v16h a0 = ldsA(As, wr * 32, 0, LDT);
    v16h a1 = ldsA(As, wr * 32 + 16, 0, LDT);
#pragma unroll
    for (int nt = 0; nt < 4; ++nt) {
      v16h b = ldsB(Bs, wc * 64 + nt * 16, 0, LDT);
      acc[0][nt] = wmma16(a0, b, acc[0][nt]);
      acc[1][nt] = wmma16(a1, b, acc[1][nt]);
    }
    __syncthreads();
  }
#pragma unroll
  for (int mt = 0; mt < 2; ++mt)
#pragma unroll
    for (int nt = 0; nt < 4; ++nt) {
      int gn = gn0 + wc * 64 + nt * 16 + nn;
      int which = gn >> 10;                    // 0=q 1=k 2=v (uniform per tile)
      int cc2 = gn & 1023;
      int h = cc2 >> 6, d = cc2 & 63;
      _Float16* dst = which == 0 ? qb : (which == 1 ? kb : vb);
      float sc = which == 0 ? 0.125f : 1.0f;   // fold D^-0.5 into Q
      float bv = bias[gn];
#pragma unroll
      for (int r = 0; r < 8; ++r) {
        int gm = gm0 + wr * 32 + mt * 16 + r + hf * 8;
        int b = gm >> 11, t = gm & 2047;
        float val = (acc[mt][nt][r] + bv) * sc;
        dst[(((size_t)(b * NH + h)) * TT + t) * HD + d] = (_Float16)val;
      }
    }
}

// ---------------- Kernel 2: causal flash attention, async double-buffered ----
__global__ __launch_bounds__(128) void csa_flash_attn(
    const _Float16* __restrict__ qb, const _Float16* __restrict__ kbuf,
    const _Float16* __restrict__ vbuf, _Float16* __restrict__ ao)
{
  const int LD = 66;
  __shared__ __align__(16) _Float16 Qs[64 * 66];
  __shared__ __align__(16) _Float16 Ks[2][64 * 66];  // [kpos][d] row-major
  __shared__ __align__(16) _Float16 Vs[2][64 * 66];  // [kpos][d] row-major
  __shared__ __align__(16) _Float16 Ps[64 * 66];

  const int bh = blockIdx.y;
  const int b = bh >> 4, h = bh & 15;
  const int qblk = blockIdx.x;
  const int qi0 = qblk * 64;
  const size_t base = (size_t)bh * TT * HD;

  const int tid = threadIdx.x;
  const int wave = tid >> 5, lane = tid & 31;
  const int hf = lane >> 4, nn = lane & 15;

  const _Float16* gK = kbuf + base;
  const _Float16* gV = vbuf + base;

  // Issue async DMA of one K/V tile pair into LDS buffer `buf`.
  auto stage = [&](int buf, int kj0) {
    unsigned kofs = (unsigned)(size_t)&Ks[buf][0];
    unsigned vofs = (unsigned)(size_t)&Vs[buf][0];
#pragma unroll
    for (int i = 0; i < 16; ++i) {
      int idx = tid + i * 128;
      int r = idx >> 5, cp = (idx & 31) * 2;
      unsigned lo = (unsigned)(r * LD + cp) * 2u;
      async_ld_b32(kofs + lo, gK + (size_t)(kj0 + r) * HD + cp);
      async_ld_b32(vofs + lo, gV + (size_t)(kj0 + r) * HD + cp);
    }
  };

#pragma unroll
  for (int i = 0; i < 16; ++i) {                   // Q tile 64x64 f16 (VGPR path)
    int idx = tid + i * 128;
    int r = idx >> 5, cp = (idx & 31) * 2;
    *(unsigned int*)&Qs[r * LD + cp] =
        *(const unsigned int*)&qb[base + (size_t)(qi0 + r) * HD + cp];
  }
  stage(0, 0);                                     // prologue DMA
  wait_async0();
  __syncthreads();

  v16h aq0 = ldsA(Qs, wave * 16, 0, LD);
  v16h aq1 = ldsA(Qs, wave * 16, 32, LD);

  v8f o[4] = {};
  float mrow[8], lrow[8];
#pragma unroll
  for (int r = 0; r < 8; ++r) { mrow[r] = -3.0e38f; lrow[r] = 0.0f; }

  for (int kb_i = 0; kb_i <= qblk; ++kb_i) {
    const int cur = kb_i & 1;
    const int kj0 = kb_i * 64;
    if (kb_i < qblk) stage(cur ^ 1, kj0 + 64);     // overlap DMA with compute
    const _Float16* Kc = Ks[cur];
    const _Float16* Vc = Vs[cur];

    v8f s[4] = {};                                 // S = Q K^T (16x64 per wave)
#pragma unroll
    for (int nt = 0; nt < 4; ++nt) {
      v16h bk0 = ldsB(Kc, nt * 16, 0, LD);
      v16h bk1 = ldsB(Kc, nt * 16, 32, LD);
      s[nt] = wmma16(aq0, bk0, s[nt]);
      s[nt] = wmma16(aq1, bk1, s[nt]);
    }

    if (kb_i == qblk) {                            // diagonal block: causal mask
#pragma unroll
      for (int nt = 0; nt < 4; ++nt)
#pragma unroll
        for (int r = 0; r < 8; ++r) {
          int qi = qi0 + wave * 16 + r + hf * 8;
          int kj = kj0 + nt * 16 + nn;
          if (kj > qi) s[nt][r] = -3.0e38f;
        }
    }

    // Online softmax; row m = r + 8*half lives in lanes {half*16 .. half*16+15}.
#pragma unroll
    for (int r = 0; r < 8; ++r) {
      float mx = fmaxf(fmaxf(s[0][r], s[1][r]), fmaxf(s[2][r], s[3][r]));
      mx = fmaxf(mx, __shfl_xor(mx, 1, 32));
      mx = fmaxf(mx, __shfl_xor(mx, 2, 32));
      mx = fmaxf(mx, __shfl_xor(mx, 4, 32));
      mx = fmaxf(mx, __shfl_xor(mx, 8, 32));
      float newm = fmaxf(mrow[r], mx);
      float alpha = __expf(mrow[r] - newm);
      mrow[r] = newm;
      float sum = 0.0f;
#pragma unroll
      for (int nt = 0; nt < 4; ++nt) {
        float p = __expf(s[nt][r] - newm);
        s[nt][r] = p;
        sum += p;
        o[nt][r] *= alpha;
      }
      sum += __shfl_xor(sum, 1, 32);
      sum += __shfl_xor(sum, 2, 32);
      sum += __shfl_xor(sum, 4, 32);
      sum += __shfl_xor(sum, 8, 32);
      lrow[r] = lrow[r] * alpha + sum;
    }

    // C-layout -> A-layout for P via this wave's private LDS rows.
#pragma unroll
    for (int nt = 0; nt < 4; ++nt)
#pragma unroll
      for (int r = 0; r < 8; ++r)
        Ps[(wave * 16 + r + hf * 8) * LD + nt * 16 + nn] = (_Float16)s[nt][r];

    v16h ap0 = ldsA(Ps, wave * 16, 0, LD);
    v16h ap1 = ldsA(Ps, wave * 16, 32, LD);
#pragma unroll
    for (int nt = 0; nt < 4; ++nt) {               // O += P V  (V row-major)
      v16h bv0 = ldsBt(Vc, nt * 16, 0, LD);
      v16h bv1 = ldsBt(Vc, nt * 16, 32, LD);
      o[nt] = wmma16(ap0, bv0, o[nt]);
      o[nt] = wmma16(ap1, bv1, o[nt]);
    }

    wait_async0();                                 // next buffer's DMA done
    __syncthreads();                               // all waves done with cur
  }

#pragma unroll
  for (int r = 0; r < 8; ++r) {
    float inv = 1.0f / lrow[r];
    int row = qi0 + wave * 16 + r + hf * 8;
#pragma unroll
    for (int nt = 0; nt < 4; ++nt) {
      int col = h * HD + nt * 16 + nn;
      ao[((size_t)(b * TT + row)) * CC + col] = (_Float16)(o[nt][r] * inv);
    }
  }
}

// ---------------- Kernel 3: out = attn_out @ Wproj + b (fp32 out) ------------
__global__ __launch_bounds__(256) void csa_proj_gemm(
    const _Float16* __restrict__ A, const float* __restrict__ W,
    const float* __restrict__ bias, float* __restrict__ out)
{
  const int LDT = 34;
  __shared__ __align__(16) _Float16 As[128 * 34];
  __shared__ __align__(16) _Float16 Bs[128 * 34];
  const int gm0 = blockIdx.y * 128;
  const int gn0 = blockIdx.x * 128;
  const int tid = threadIdx.x;
  const int wave = tid >> 5, lane = tid & 31;
  const int wr = wave >> 1, wc = wave & 1;
  const int hf = lane >> 4, nn = lane & 15;

  v8f acc[2][4] = {};
  for (int k0 = 0; k0 < CC; k0 += 32) {
    if (k0 + 32 < CC) {
      __builtin_prefetch(&A[(size_t)(gm0 + (tid >> 4)) * CC + k0 + 32 + (tid & 15) * 2], 0, 0);
      __builtin_prefetch(&W[(size_t)(k0 + 32 + (tid >> 7)) * CC + gn0 + (tid & 127)], 0, 0);
    }
#pragma unroll
    for (int i = 0; i < 8; ++i) {                  // A tile: f16 pairs
      int idx = tid + i * 256;
      int r = idx >> 4, cp = (idx & 15) * 2;
      *(unsigned int*)&As[r * LDT + cp] =
          *(const unsigned int*)&A[(size_t)(gm0 + r) * CC + k0 + cp];
    }
#pragma unroll
    for (int i = 0; i < 16; ++i) {
      int idx = tid + i * 256;
      int rk = idx >> 7, cn = idx & 127;
      Bs[cn * LDT + rk] = (_Float16)W[(size_t)(k0 + rk) * CC + gn0 + cn];
    }
    __syncthreads();
    v16h a0 = ldsA(As, wr * 32, 0, LDT);
    v16h a1 = ldsA(As, wr * 32 + 16, 0, LDT);
#pragma unroll
    for (int nt = 0; nt < 4; ++nt) {
      v16h b = ldsB(Bs, wc * 64 + nt * 16, 0, LDT);
      acc[0][nt] = wmma16(a0, b, acc[0][nt]);
      acc[1][nt] = wmma16(a1, b, acc[1][nt]);
    }
    __syncthreads();
  }
#pragma unroll
  for (int mt = 0; mt < 2; ++mt)
#pragma unroll
    for (int nt = 0; nt < 4; ++nt) {
      int gn = gn0 + wc * 64 + nt * 16 + nn;
      float bv = bias[gn];
#pragma unroll
      for (int r = 0; r < 8; ++r) {
        int gm = gm0 + wr * 32 + mt * 16 + r + hf * 8;
        out[(size_t)gm * CC + gn] = acc[mt][nt][r] + bv;
      }
    }
}

extern "C" void kernel_launch(void* const* d_in, const int* in_sizes, int n_in,
                              void* d_out, int out_size, void* d_ws, size_t ws_size,
                              hipStream_t stream) {
  const float* x     = (const float*)d_in[0];
  const float* Wqkv  = (const float*)d_in[1];
  const float* bqkv  = (const float*)d_in[2];
  const float* Wproj = (const float*)d_in[3];
  const float* bproj = (const float*)d_in[4];
  float* out = (float*)d_out;

  char* ws = (char*)d_ws;
  const size_t SZ = (size_t)4 * NH * TT * HD * sizeof(_Float16);  // 16 MB each
  _Float16* qb = (_Float16*)(ws);
  _Float16* kb = (_Float16*)(ws + SZ);
  _Float16* vb = (_Float16*)(ws + 2 * SZ);
  _Float16* ao = (_Float16*)(ws + 3 * SZ);          // 64 MB workspace total

  csa_qkv_gemm<<<dim3(24, 64), 256, 0, stream>>>(x, Wqkv, bqkv, qb, kb, vb);
  csa_flash_attn<<<dim3(32, 64), 128, 0, stream>>>(qb, kb, vb, ao);
  csa_proj_gemm<<<dim3(8, 64), 256, 0, stream>>>(ao, Wproj, bproj, out);
}
